// VimSSM_5059471475436
// MI455X (gfx1250) — compile-verified, matrix operand order checked by reading, not run on hardware
//
#include <hip/hip_runtime.h>
#include <hip/hip_bf16.h>
#include <math.h>

// ---------------- Problem constants (from reference) ----------------
#define DM    1024            // D_MODEL
#define DI    2048            // D_INNER
#define LSEQ  2048
#define BSZ   4
#define BL    (BSZ * LSEQ)    // 8192 rows (b,l flattened)
#define DTR   64              // DT_RANK
#define DST   16              // D_STATE
#define XD    (DTR + 2*DST)   // 96
#define CL    64              // scan chunk length
#define NC    (LSEQ / CL)     // 32 chunks

// ---------------- WMMA types ----------------
typedef __attribute__((ext_vector_type(16))) __bf16 v16bf;
typedef __attribute__((ext_vector_type(8)))  float  v8f;

union Frag {
    v16bf v;
    uint4 q[2];
};

// ---------------- bf16 helpers (RNE) ----------------
__device__ __forceinline__ unsigned short f2bf(float f) {
    unsigned int u = __float_as_uint(f);
    u += 0x7FFFu + ((u >> 16) & 1u);
    return (unsigned short)(u >> 16);
}
__device__ __forceinline__ float bf2f(unsigned short h) {
    return __uint_as_float(((unsigned int)h) << 16);
}

// ---------------- f32 -> bf16 cast ----------------
__global__ void cast_bf16_kernel(const float* __restrict__ src,
                                 unsigned short* __restrict__ dst, size_t n) {
    size_t i = (size_t)blockIdx.x * blockDim.x + threadIdx.x;
    if (i < n) dst[i] = f2bf(src[i]);
}

// ---------------- WMMA GEMM: C[M,N] = A[M,K] * Bw[N,K]^T ----------------
// Register-tiled: each wave owns a 64x32 tile of C = 4 (M) x 2 (N) fragments,
// 8 accumulators, 8 v_wmma per K-step fed by 12 b128 loads (~21 flop/byte).
// mode 0: store f32; mode 1: store bf16; mode 2: store f32 softplus(c + bias[n]).
__global__ void wmma_gemm_kernel(const unsigned short* __restrict__ A, int lda,
                                 const unsigned short* __restrict__ Bw, int ldb,
                                 void* __restrict__ Cout, int ldc,
                                 int M, int N, int K, int mode,
                                 const float* __restrict__ bias) {
    int wave = (int)((blockIdx.x * (size_t)blockDim.x + threadIdx.x) >> 5);
    int lane = threadIdx.x & 31;
    int nwt  = N >> 5;                       // 32-wide wave tiles along N
    int mw   = wave / nwt;                   // 64-tall wave tile along M
    int nw   = wave - mw * nwt;
    if (mw >= (M >> 6)) return;              // wave-uniform: EXEC stays all-1s

    int half = lane >> 4;
    int r    = lane & 15;
    int koff = 8 * half;                     // bf16 fragment: K = {8h..8h+7, 16+8h..16+8h+7}

    const unsigned short* arow[4];
    const unsigned short* brow[2];
    #pragma unroll
    for (int i = 0; i < 4; ++i)
        arow[i] = A + (size_t)(mw * 64 + i * 16 + r) * (size_t)lda;
    #pragma unroll
    for (int j = 0; j < 2; ++j)
        brow[j] = Bw + (size_t)(nw * 32 + j * 16 + r) * (size_t)ldb;

    v8f acc[4][2] = {};
    for (int k = 0; k < K; k += 32) {
        // pull next K-slab toward the caches (global_prefetch_b8, speculative)
        __builtin_prefetch(arow[0] + k + 32, 0, 1);
        __builtin_prefetch(brow[0] + k + 32, 0, 1);

        Frag a[4], b[2];
        #pragma unroll
        for (int i = 0; i < 4; ++i) {
            a[i].q[0] = *(const uint4*)(arow[i] + k + koff);
            a[i].q[1] = *(const uint4*)(arow[i] + k + 16 + koff);
        }
        #pragma unroll
        for (int j = 0; j < 2; ++j) {
            b[j].q[0] = *(const uint4*)(brow[j] + k + koff);
            b[j].q[1] = *(const uint4*)(brow[j] + k + 16 + koff);
        }
        #pragma unroll
        for (int i = 0; i < 4; ++i)
            #pragma unroll
            for (int j = 0; j < 2; ++j)
                acc[i][j] = __builtin_amdgcn_wmma_f32_16x16x32_bf16(
                    false, a[i].v, false, b[j].v, (short)0, acc[i][j], false, false);
    }

    // C layout: lanes 0-15 rows 0..7 in acc[.][.][0..7]; lanes 16-31 rows 8..15
    #pragma unroll
    for (int i = 0; i < 4; ++i) {
        int row0 = mw * 64 + i * 16 + 8 * half;
        #pragma unroll
        for (int j = 0; j < 2; ++j) {
            int col = nw * 32 + j * 16 + r;
            if (mode == 0) {
                float* C = (float*)Cout;
                #pragma unroll
                for (int t = 0; t < 8; ++t)
                    C[(size_t)(row0 + t) * (size_t)ldc + col] = acc[i][j][t];
            } else if (mode == 1) {
                unsigned short* C = (unsigned short*)Cout;
                #pragma unroll
                for (int t = 0; t < 8; ++t)
                    C[(size_t)(row0 + t) * (size_t)ldc + col] = f2bf(acc[i][j][t]);
            } else {
                float* C = (float*)Cout;
                float bv = bias[col];
                #pragma unroll
                for (int t = 0; t < 8; ++t) {
                    float x = acc[i][j][t] + bv;
                    C[(size_t)(row0 + t) * (size_t)ldc + col] =
                        (x > 20.f) ? x : log1pf(__expf(x));   // softplus
                }
            }
        }
    }
}

// ---------------- Depthwise causal conv (k=4) + SiLU ----------------
__global__ void conv_silu_kernel(const unsigned short* __restrict__ xzb,
                                 const float* __restrict__ convw,
                                 const float* __restrict__ convb,
                                 unsigned short* __restrict__ ub) {
    size_t idx = (size_t)blockIdx.x * blockDim.x + threadIdx.x;
    if (idx >= (size_t)BL * DI) return;
    int d  = (int)(idx % DI);
    size_t bl = idx / DI;
    int l  = (int)(bl % LSEQ);
    size_t brow = bl - l;                          // b*LSEQ
    float acc = convb[d];
    #pragma unroll
    for (int j = 0; j < 4; ++j) {
        int ls = l - 3 + j;
        if (ls >= 0)
            acc += convw[d * 4 + j] * bf2f(xzb[(brow + ls) * (size_t)(2 * DI) + d]);
    }
    float s = acc / (1.f + __expf(-acc));          // silu
    ub[idx] = f2bf(s);
}

// ---------------- Chunked selective scan ----------------
// h_t = exp(dt*a)*h_{t-1} + dt*u*B_t[n] is a scalar linear recurrence, so we
// run 32 chunks of 64 steps in parallel, then stitch initial states.

// Phase 1: per-chunk local scan with h0=0 -> F[b,c,d,n]; also Ssum = sum(dt).
__global__ void scan_phase1_kernel(const float* __restrict__ delta,
                                   const unsigned short* __restrict__ ub,
                                   const unsigned short* __restrict__ xdblb,
                                   const float* __restrict__ A_log,
                                   float* __restrict__ F,
                                   float* __restrict__ Ssum) {
    const int dpb = 256;
    int bc   = blockIdx.x / (DI / dpb);            // b*NC + c
    int dblk = blockIdx.x % (DI / dpb);
    int b = bc / NC, c = bc % NC;
    int d = dblk * dpb + threadIdx.x;

    float a[DST], h[DST];
    #pragma unroll
    for (int n = 0; n < DST; ++n) {
        a[n] = -__expf(A_log[d * DST + n]);
        h[n] = 0.f;
    }
    float dtsum = 0.f;
    __shared__ float sB[DST];

    for (int l = c * CL; l < c * CL + CL; ++l) {
        size_t blo = (size_t)b * LSEQ + l;
        if (threadIdx.x < DST)
            sB[threadIdx.x] = bf2f(xdblb[blo * XD + DTR + threadIdx.x]);
        __syncthreads();
        float dt = delta[blo * DI + d];
        float du = dt * bf2f(ub[blo * DI + d]);
        dtsum += dt;
        #pragma unroll
        for (int n = 0; n < DST; ++n)
            h[n] = __expf(dt * a[n]) * h[n] + du * sB[n];
        __syncthreads();
    }
    size_t fo = ((size_t)bc * DI + d) * DST;
    #pragma unroll
    for (int n = 0; n < DST; ++n) F[fo + n] = h[n];
    Ssum[(size_t)bc * DI + d] = dtsum;
}

// Phase 2: serial chain over the 32 chunks, parallel over (b,d,n) = 131072
// threads. Rewrites F in place: F[b,c,d,n] becomes chunk c's INITIAL state.
__global__ void scan_phase2_kernel(const float* __restrict__ A_log,
                                   const float* __restrict__ Ssum,
                                   float* __restrict__ F) {
    size_t idx = (size_t)blockIdx.x * blockDim.x + threadIdx.x;
    if (idx >= (size_t)BSZ * DI * DST) return;
    int n = (int)(idx % DST);
    size_t t = idx / DST;
    int d = (int)(t % DI);
    int b = (int)(t / DI);
    float a = -__expf(A_log[d * DST + n]);
    float h = 0.f;
    for (int c = 0; c < NC; ++c) {
        size_t bc = (size_t)b * NC + c;
        size_t fo = (bc * DI + d) * DST + n;
        float f = F[fo];                           // chunk-local final (h0=0)
        float g = __expf(a * Ssum[bc * DI + d]);   // chunk decay product
        F[fo] = h;                                 // becomes chunk initial state
        h = g * h + f;
    }
}

// Phase 3: replay chunks in parallel from corrected initial states; fuse
// y = (h.C + D*u) * silu(z) and emit bf16 y for the output GEMM.
__global__ void scan_phase3_kernel(const float* __restrict__ delta,
                                   const unsigned short* __restrict__ ub,
                                   const unsigned short* __restrict__ xdblb,
                                   const unsigned short* __restrict__ xzb,
                                   const float* __restrict__ A_log,
                                   const float* __restrict__ Dp,
                                   const float* __restrict__ F,
                                   unsigned short* __restrict__ yb) {
    const int dpb = 256;
    int bc   = blockIdx.x / (DI / dpb);
    int dblk = blockIdx.x % (DI / dpb);
    int b = bc / NC, c = bc % NC;
    int d = dblk * dpb + threadIdx.x;

    float a[DST], h[DST];
    size_t fo = ((size_t)bc * DI + d) * DST;
    #pragma unroll
    for (int n = 0; n < DST; ++n) {
        a[n] = -__expf(A_log[d * DST + n]);
        h[n] = F[fo + n];                          // corrected initial state
    }
    float Dd = Dp[d];
    __shared__ float sB[DST];
    __shared__ float sC[DST];

    for (int l = c * CL; l < c * CL + CL; ++l) {
        size_t blo = (size_t)b * LSEQ + l;
        if (threadIdx.x < 2 * DST) {
            float v = bf2f(xdblb[blo * XD + DTR + threadIdx.x]);
            if (threadIdx.x < DST) sB[threadIdx.x] = v;
            else                   sC[threadIdx.x - DST] = v;
        }
        __syncthreads();
        float dt = delta[blo * DI + d];
        float ut = bf2f(ub[blo * DI + d]);
        float du = dt * ut;
        float y  = 0.f;
        #pragma unroll
        for (int n = 0; n < DST; ++n) {
            h[n] = __expf(dt * a[n]) * h[n] + du * sB[n];
            y += h[n] * sC[n];
        }
        y += Dd * ut;
        float zz = bf2f(xzb[blo * (size_t)(2 * DI) + DI + d]);
        y *= zz / (1.f + __expf(-zz));             // * silu(z)
        yb[blo * DI + d] = f2bf(y);
        __syncthreads();
    }
}

// ---------------- Orchestration ----------------
extern "C" void kernel_launch(void* const* d_in, const int* in_sizes, int n_in,
                              void* d_out, int out_size, void* d_ws, size_t ws_size,
                              hipStream_t stream) {
    const float* x      = (const float*)d_in[0];   // [4,2048,1024]
    const float* W_in   = (const float*)d_in[1];   // [4096,1024]
    const float* conv_w = (const float*)d_in[2];   // [2048,1,4]
    const float* conv_b = (const float*)d_in[3];   // [2048]
    const float* W_xp   = (const float*)d_in[4];   // [96,2048]
    const float* W_dt   = (const float*)d_in[5];   // [2048,64]
    const float* b_dt   = (const float*)d_in[6];   // [2048]
    const float* A_log  = (const float*)d_in[7];   // [2048,16]
    const float* Dp     = (const float*)d_in[8];   // [2048]
    const float* W_out  = (const float*)d_in[9];   // [1024,2048]
    float* out = (float*)d_out;                    // [4,2048,1024]

    char* ws = (char*)d_ws;
    size_t off = 0;
    auto take = [&](size_t bytes) -> char* {
        char* p = ws + off;
        off = (off + bytes + 255) & ~(size_t)255;
        return p;
    };
    unsigned short* xb    = (unsigned short*)take((size_t)BL * DM * 2);
    unsigned short* winb  = (unsigned short*)take((size_t)2 * DI * DM * 2);
    unsigned short* xzb   = (unsigned short*)take((size_t)BL * 2 * DI * 2);
    unsigned short* ub    = (unsigned short*)take((size_t)BL * DI * 2);
    unsigned short* wxpb  = (unsigned short*)take((size_t)XD * DI * 2);
    unsigned short* xdblb = (unsigned short*)take((size_t)BL * XD * 2);
    unsigned short* wdtb  = (unsigned short*)take((size_t)DI * DTR * 2);
    float*          dlt   = (float*)take((size_t)BL * DI * 4);
    unsigned short* ybuf  = (unsigned short*)take((size_t)BL * DI * 2);
    unsigned short* woutb = (unsigned short*)take((size_t)DM * DI * 2);
    float*          Fbuf  = (float*)take((size_t)BSZ * NC * DI * DST * 4);
    float*          Sbuf  = (float*)take((size_t)BSZ * NC * DI * 4);

    auto cast = [&](const float* s, unsigned short* d, size_t n) {
        cast_bf16_kernel<<<(unsigned)((n + 255) / 256), 256, 0, stream>>>(s, d, n);
    };
    cast(x,     xb,    (size_t)BL * DM);
    cast(W_in,  winb,  (size_t)2 * DI * DM);
    cast(W_xp,  wxpb,  (size_t)XD * DI);
    cast(W_dt,  wdtb,  (size_t)DI * DTR);
    cast(W_out, woutb, (size_t)DM * DI);

    auto gemm = [&](const unsigned short* A, int lda, const unsigned short* B, int ldb,
                    void* C, int ldc, int M, int N, int K, int mode, const float* bias) {
        unsigned blocks = (unsigned)(((size_t)(M / 64) * (N / 32)) / 8);
        wmma_gemm_kernel<<<blocks, 256, 0, stream>>>(A, lda, B, ldb, C, ldc,
                                                     M, N, K, mode, bias);
    };

    // 1) xz = x @ W_in^T            (8192 x 4096, K=1024) -> bf16
    gemm(xb, DM, winb, DM, xzb, 2 * DI, BL, 2 * DI, DM, 1, nullptr);
    // 2) depthwise conv + silu -> u (bf16)
    {
        size_t n = (size_t)BL * DI;
        conv_silu_kernel<<<(unsigned)((n + 255) / 256), 256, 0, stream>>>(xzb, conv_w, conv_b, ub);
    }
    // 3) x_dbl = u @ W_xproj^T      (8192 x 96, K=2048) -> bf16
    gemm(ub, DI, wxpb, DI, xdblb, XD, BL, XD, DI, 1, nullptr);
    // 4) delta = softplus(delta_raw @ W_dt^T + b_dt)  (A strided in x_dbl, lda=96, K=64)
    gemm(xdblb, XD, wdtb, DTR, dlt, DI, BL, DI, DTR, 2, b_dt);
    // 5) chunked parallel scan (32 chunks x 64 steps), fused gating -> bf16 y
    scan_phase1_kernel<<<BSZ * NC * (DI / 256), 256, 0, stream>>>(dlt, ub, xdblb, A_log, Fbuf, Sbuf);
    scan_phase2_kernel<<<(BSZ * DI * DST) / 256, 256, 0, stream>>>(A_log, Sbuf, Fbuf);
    scan_phase3_kernel<<<BSZ * NC * (DI / 256), 256, 0, stream>>>(dlt, ub, xdblb, xzb, A_log, Dp,
                                                                  Fbuf, ybuf);
    // 6) out = y @ W_out^T          (8192 x 1024, K=2048) -> f32 d_out
    gemm(ybuf, DI, woutb, DI, out, DM, BL, DM, DI, 0, nullptr);
}
